// LlamaAttention_39393440039015
// MI455X (gfx1250) — matrix operand order repaired and checked by reading, not compiled, not run on hardware
//
#include <hip/hip_runtime.h>
#include <hip/hip_bf16.h>

typedef __attribute__((ext_vector_type(16))) _Float16 v16h;
typedef __attribute__((ext_vector_type(8)))  _Float16 v8h;
typedef __attribute__((ext_vector_type(4)))  _Float16 v4h;
typedef __attribute__((ext_vector_type(8)))  float    v8f;

#define S_LEN   2048
#define HID2    8192
#define NH      32
#define NKV     8
#define HD      128
#define QKV_COLS 6144   // 4096 q + 1024 k + 1024 v

// Load a v16h fragment from two 16-byte-aligned 8-half chunks.
static __device__ __forceinline__ v16h ld_frag16(const _Float16* p0, const _Float16* p1) {
    v16h r;
    *(v8h*)&r       = *(const v8h*)p0;
    *((v8h*)&r + 1) = *(const v8h*)p1;
    return r;
}

static __device__ __forceinline__ v8f wmma_f16(v16h a, v16h b, v8f c) {
    return __builtin_amdgcn_wmma_f32_16x16x32_f16(false, a, false, b, (short)0, c,
                                                  false, false);
}

// ---------------------------------------------------------------------------
// fp32 -> f16 bulk convert (4 elems/thread)
__global__ __launch_bounds__(256)
void f32_to_f16_kernel(const float* __restrict__ src, _Float16* __restrict__ dst, int n) {
    int i = (blockIdx.x * 256 + threadIdx.x) * 4;
    if (i >= n) return;
    float4 f = *(const float4*)(src + i);
    v4h h;
    h[0] = (_Float16)f.x; h[1] = (_Float16)f.y;
    h[2] = (_Float16)f.z; h[3] = (_Float16)f.w;
    *(v4h*)(dst + i) = h;
}

// ---------------------------------------------------------------------------
// GEMM: C[m, n] = sum_k A16[m,k] * W32[n,k]  (A f16 row-major MxK, W fp32 NxK)
// Tile 128x128x32, 256 threads = 8 waves (4 M-waves x 2 N-waves),
// each wave: 32x64 output = 2x4 WMMA accumulators.
__global__ __launch_bounds__(256)
void gemm_f16w32_kernel(const _Float16* __restrict__ A,
                        const float* __restrict__ W,
                        float* __restrict__ C,
                        int K, int ldc) {
    __shared__ _Float16 As[128][40];   // 32 + 8 pad halves (80 B rows)
    __shared__ _Float16 Bs[128][40];

    const int tid  = threadIdx.x;
    const int lane = tid & 31;
    const int wv   = tid >> 5;
    const int wm   = wv >> 1;          // 0..3
    const int wn   = wv & 1;           // 0..1
    const int l16  = lane & 15;
    const int hsel = lane >> 4;
    const int mbase = blockIdx.y * 128;
    const int nbase = blockIdx.x * 128;

    const v8f zero8 = {0.f,0.f,0.f,0.f,0.f,0.f,0.f,0.f};
    v8f acc[2][4];
    #pragma unroll
    for (int mi = 0; mi < 2; ++mi)
        #pragma unroll
        for (int ni = 0; ni < 4; ++ni) acc[mi][ni] = zero8;

    const int lrow  = tid >> 1;        // 0..127
    const int lkoff = (tid & 1) * 16;  // 0 or 16

    for (int k0 = 0; k0 < K; k0 += 32) {
        __syncthreads();
        // stage A tile (f16, straight copy)
        const _Float16* ap = A + (size_t)(mbase + lrow) * K + k0 + lkoff;
        *(v8h*)&As[lrow][lkoff]     = *(const v8h*)ap;
        *(v8h*)&As[lrow][lkoff + 8] = *(const v8h*)(ap + 8);
        // stage W tile (fp32 -> f16 convert on the fly)
        const float* wp = W + (size_t)(nbase + lrow) * K + k0 + lkoff;
        {
            float4 f0 = *(const float4*)(wp + 0);
            float4 f1 = *(const float4*)(wp + 4);
            float4 f2 = *(const float4*)(wp + 8);
            float4 f3 = *(const float4*)(wp + 12);
            v8h b0, b1;
            b0[0]=(_Float16)f0.x; b0[1]=(_Float16)f0.y; b0[2]=(_Float16)f0.z; b0[3]=(_Float16)f0.w;
            b0[4]=(_Float16)f1.x; b0[5]=(_Float16)f1.y; b0[6]=(_Float16)f1.z; b0[7]=(_Float16)f1.w;
            b1[0]=(_Float16)f2.x; b1[1]=(_Float16)f2.y; b1[2]=(_Float16)f2.z; b1[3]=(_Float16)f2.w;
            b1[4]=(_Float16)f3.x; b1[5]=(_Float16)f3.y; b1[6]=(_Float16)f3.z; b1[7]=(_Float16)f3.w;
            *(v8h*)&Bs[lrow][lkoff]     = b0;
            *(v8h*)&Bs[lrow][lkoff + 8] = b1;
        }
        __syncthreads();

        #pragma unroll
        for (int mi = 0; mi < 2; ++mi) {
            const int arow = wm*32 + mi*16 + l16;
            const int akb  = hsel * 8;
            v16h a = ld_frag16(&As[arow][akb], &As[arow][akb + 16]);
            #pragma unroll
            for (int ni = 0; ni < 4; ++ni) {
                const int bcol = wn*64 + ni*16 + l16;
                const int bkb  = hsel * 8;
                v16h b = ld_frag16(&Bs[bcol][bkb], &Bs[bcol][bkb + 16]);
                acc[mi][ni] = wmma_f16(a, b, acc[mi][ni]);
            }
        }
    }

    #pragma unroll
    for (int mi = 0; mi < 2; ++mi)
        #pragma unroll
        for (int ni = 0; ni < 4; ++ni) {
            const int gm = mbase + wm*32 + mi*16 + 8*hsel;
            const int gn = nbase + wn*64 + ni*16 + l16;
            #pragma unroll
            for (int v = 0; v < 8; ++v)
                C[(size_t)(gm + v) * ldc + gn] = acc[mi][ni][v];
        }
}

// ---------------------------------------------------------------------------
// RoPE on q/k, prescale q by 1/sqrt(HD), write head-major f16.
// One thread per (s, head-or-kv, pair i).
__global__ __launch_bounds__(256)
void rope_qk_kernel(const float* __restrict__ qkv, const float* __restrict__ freqs,
                    _Float16* __restrict__ q16, _Float16* __restrict__ k16) {
    int idx = blockIdx.x * 256 + threadIdx.x;   // [0, S*40*64)
    int s  = idx / 2560;
    int r  = idx - s * 2560;
    int hh = r >> 6;
    int i  = r & 63;
    float c  = freqs[s * 128 + i * 2 + 0];
    float sn = freqs[s * 128 + i * 2 + 1];
    if (hh < NH) {
        const float* p = qkv + (size_t)s * QKV_COLS + hh * HD + i * 2;
        float x0 = p[0], x1 = p[1];
        const float sc = 0.08838834764831845f;  // 1/sqrt(128)
        _Float16* o = q16 + ((size_t)hh * S_LEN + s) * HD + i * 2;
        o[0] = (_Float16)((x0 * c - x1 * sn) * sc);
        o[1] = (_Float16)((x1 * c + x0 * sn) * sc);
    } else {
        int kv = hh - NH;
        const float* p = qkv + (size_t)s * QKV_COLS + 4096 + kv * HD + i * 2;
        float x0 = p[0], x1 = p[1];
        _Float16* o = k16 + ((size_t)kv * S_LEN + s) * HD + i * 2;
        o[0] = (_Float16)(x0 * c - x1 * sn);
        o[1] = (_Float16)(x1 * c + x0 * sn);
    }
}

// v: straight convert to head-major f16 (4 elems/thread)
__global__ __launch_bounds__(256)
void conv_v_kernel(const float* __restrict__ qkv, _Float16* __restrict__ vout) {
    int idx4 = (blockIdx.x * 256 + threadIdx.x) * 4;   // [0, S*1024)
    int s = idx4 >> 10;
    int r = idx4 & 1023;
    int kv = r >> 7;
    int d  = r & 127;
    float4 f = *(const float4*)(qkv + (size_t)s * QKV_COLS + 5120 + r);
    v4h h;
    h[0]=(_Float16)f.x; h[1]=(_Float16)f.y; h[2]=(_Float16)f.z; h[3]=(_Float16)f.w;
    *(v4h*)(vout + ((size_t)kv * S_LEN + s) * HD + d) = h;
}

// ---------------------------------------------------------------------------
// Flash attention per (head, 64-query tile). 128 threads = 4 waves,
// each wave owns 16 query rows. Keys: causal k0 blocks of 64 + 2 diag extras.
__global__ __launch_bounds__(128)
void attn_kernel(const _Float16* __restrict__ q16, const _Float16* __restrict__ k16,
                 const _Float16* __restrict__ v16p, const float* __restrict__ ckp,
                 const float* __restrict__ cvp, _Float16* __restrict__ attn16) {
    __shared__ _Float16 Kt[64][136];      // [key][d], padded
    __shared__ _Float16 Vt[128][72];      // [d][key] (transposed), padded
    __shared__ _Float16 Pt[4][16][72];    // per-wave probs [row][key]
    __shared__ float    exs[4][2][16];    // per-wave extra-key logits

    const int h    = blockIdx.y;
    const int qb   = blockIdx.x;
    const int tid  = threadIdx.x;
    const int lane = tid & 31;
    const int wv   = tid >> 5;
    const int l16  = lane & 15;
    const int hsel = lane >> 4;
    const int qrow0 = qb * 64 + wv * 16;
    const int kvh   = h >> 2;             // GQA: 4 q-heads per kv-head

    const _Float16* qh  = q16 + (size_t)h * S_LEN * HD;
    const float* k0p = ckp + (size_t)h * S_LEN * HD;              // cache_k_prev[0]
    const float* ek0 = ckp + ((size_t)NH + h) * S_LEN * HD;       // cache_k_prev[1]
    const float* v0p = cvp + (size_t)h * S_LEN * HD;              // cache_v_prev[0]
    const float* ev0 = cvp + ((size_t)NH + h) * S_LEN * HD;       // cache_v_prev[1]
    const _Float16* kN = k16 + (size_t)kvh * S_LEN * HD;          // new k (roped)
    const _Float16* vN = v16p + (size_t)kvh * S_LEN * HD;         // new v

    // Q fragments (A-matrix, 16x32 each over 4 k-chunks); q is pre-scaled.
    v16h aq[4];
    {
        const _Float16* qp = qh + (size_t)(qrow0 + l16) * HD;
        #pragma unroll
        for (int kc = 0; kc < 4; ++kc)
            aq[kc] = ld_frag16(qp + kc*32 + hsel*8, qp + kc*32 + 16 + hsel*8);
    }

    // Diagonal extras: per-row 128-dot with ek0 / new-k (2 lanes per row).
    {
        int rg = qrow0 + l16;
        const _Float16* qp  = qh + (size_t)rg * HD + hsel * 64;
        const float*    e0p = ek0 + (size_t)rg * HD + hsel * 64;
        const _Float16* e1p = kN  + (size_t)rg * HD + hsel * 64;
        float s0 = 0.f, s1 = 0.f;
        #pragma unroll 8
        for (int d = 0; d < 64; ++d) {
            float qd = (float)qp[d];
            s0 += qd * e0p[d];
            s1 += qd * (float)e1p[d];
        }
        s0 += __shfl_xor(s0, 16, 32);
        s1 += __shfl_xor(s1, 16, 32);
        if (hsel == 0) { exs[wv][0][l16] = s0; exs[wv][1][l16] = s1; }
    }

    const v8f zero8 = {0.f,0.f,0.f,0.f,0.f,0.f,0.f,0.f};
    float mrun[8], lrun[8];
    v8f o[8];
    #pragma unroll
    for (int v = 0; v < 8; ++v) { mrun[v] = -1e30f; lrun[v] = 0.f; o[v] = zero8; }

    for (int kb = 0; kb <= qb; ++kb) {
        __syncthreads();
        {   // stage K tile (f16) and V tile (f16, transposed)
            const int row  = tid >> 1;
            const int doff = (tid & 1) * 64;
            const float* kp = k0p + (size_t)(kb*64 + row) * HD + doff;
            #pragma unroll
            for (int j = 0; j < 64; j += 8) {
                float4 f0 = *(const float4*)(kp + j);
                float4 f1 = *(const float4*)(kp + j + 4);
                v8h hh8;
                hh8[0]=(_Float16)f0.x; hh8[1]=(_Float16)f0.y; hh8[2]=(_Float16)f0.z; hh8[3]=(_Float16)f0.w;
                hh8[4]=(_Float16)f1.x; hh8[5]=(_Float16)f1.y; hh8[6]=(_Float16)f1.z; hh8[7]=(_Float16)f1.w;
                *(v8h*)&Kt[row][doff + j] = hh8;
            }
            const float* vp = v0p + (size_t)(kb*64 + row) * HD + doff;
            #pragma unroll
            for (int j = 0; j < 64; j += 4) {
                float4 f = *(const float4*)(vp + j);
                Vt[doff + j + 0][row] = (_Float16)f.x;
                Vt[doff + j + 1][row] = (_Float16)f.y;
                Vt[doff + j + 2][row] = (_Float16)f.z;
                Vt[doff + j + 3][row] = (_Float16)f.w;
            }
        }
        __syncthreads();

        // scores: 16 rows x 64 keys
        v8f sc[4];
        #pragma unroll
        for (int ni = 0; ni < 4; ++ni) {
            sc[ni] = zero8;
            #pragma unroll
            for (int kc = 0; kc < 4; ++kc) {
                const _Float16* kr = &Kt[ni*16 + l16][kc*32 + hsel*8];
                v16h b = ld_frag16(kr, kr + 16);
                sc[ni] = wmma_f16(aq[kc], b, sc[ni]);
            }
        }
        if (kb == qb) {   // causal mask inside diagonal block
            #pragma unroll
            for (int ni = 0; ni < 4; ++ni) {
                int key = kb*64 + ni*16 + l16;
                #pragma unroll
                for (int v = 0; v < 8; ++v)
                    if (key > qrow0 + v + 8*hsel) sc[ni][v] = -1e9f;
            }
        }

        // online softmax (row = vgpr v + 8*hsel, reduce across 16-lane halves)
        float alpha[8];
        float p[4][8];
        #pragma unroll
        for (int v = 0; v < 8; ++v) {
            float m = fmaxf(fmaxf(sc[0][v], sc[1][v]), fmaxf(sc[2][v], sc[3][v]));
            #pragma unroll
            for (int x = 1; x < 16; x <<= 1) m = fmaxf(m, __shfl_xor(m, x, 32));
            float mn = fmaxf(mrun[v], m);
            float a  = __expf(mrun[v] - mn);
            float s  = 0.f;
            #pragma unroll
            for (int ni = 0; ni < 4; ++ni) {
                float pv = __expf(sc[ni][v] - mn);
                p[ni][v] = pv;
                s += pv;
            }
            #pragma unroll
            for (int x = 1; x < 16; x <<= 1) s += __shfl_xor(s, x, 32);
            lrun[v] = lrun[v] * a + s;
            mrun[v] = mn;
            alpha[v] = a;
        }
        #pragma unroll
        for (int dj = 0; dj < 8; ++dj)
            #pragma unroll
            for (int v = 0; v < 8; ++v) o[dj][v] *= alpha[v];

        // re-layout P: C-layout -> LDS -> A-fragments
        #pragma unroll
        for (int ni = 0; ni < 4; ++ni)
            #pragma unroll
            for (int v = 0; v < 8; ++v)
                Pt[wv][v + 8*hsel][ni*16 + l16] = (_Float16)p[ni][v];

        #pragma unroll
        for (int kc = 0; kc < 2; ++kc) {
            const _Float16* pr = &Pt[wv][l16][kc*32 + hsel*8];
            v16h ap = ld_frag16(pr, pr + 16);
            #pragma unroll
            for (int dj = 0; dj < 8; ++dj) {
                const _Float16* vr = &Vt[dj*16 + l16][kc*32 + hsel*8];
                v16h bv = ld_frag16(vr, vr + 16);
                o[dj] = wmma_f16(ap, bv, o[dj]);
            }
        }
    }

    // merge the two diagonal extras, normalize, store f16 [s][NH*HD]
    float af[8], w0[8], w1[8], lf[8];
    #pragma unroll
    for (int v = 0; v < 8; ++v) {
        int rloc = v + 8*hsel;
        float x0 = exs[wv][0][rloc];
        float x1 = exs[wv][1][rloc];
        float mf = fmaxf(mrun[v], fmaxf(x0, x1));
        af[v] = __expf(mrun[v] - mf);
        w0[v] = __expf(x0 - mf);
        w1[v] = __expf(x1 - mf);
        lf[v] = lrun[v] * af[v] + w0[v] + w1[v];
    }
    #pragma unroll
    for (int dj = 0; dj < 8; ++dj) {
        #pragma unroll
        for (int v = 0; v < 8; ++v) {
            int rg = qrow0 + v + 8*hsel;
            int d  = dj*16 + l16;
            float ov = o[dj][v] * af[v]
                     + w0[v] * ev0[(size_t)rg * HD + d]
                     + w1[v] * (float)vN[(size_t)rg * HD + d];
            attn16[(size_t)rg * (NH*HD) + h * HD + d] = (_Float16)(ov / lf[v]);
        }
    }
}

// ---------------------------------------------------------------------------
extern "C" void kernel_launch(void* const* d_in, const int* in_sizes, int n_in,
                              void* d_out, int out_size, void* d_ws, size_t ws_size,
                              hipStream_t stream) {
    const float* hs  = (const float*)d_in[0];
    // d_in[1] attention_mask: causal, reconstructed in-kernel
    const float* fr  = (const float*)d_in[2];
    const float* ckp = (const float*)d_in[3];
    const float* cvp = (const float*)d_in[4];
    const float* Wq  = (const float*)d_in[5];
    const float* Wk  = (const float*)d_in[6];
    const float* Wv  = (const float*)d_in[7];
    const float* Wo  = (const float*)d_in[8];
    float* out = (float*)d_out;
    (void)in_sizes; (void)n_in; (void)out_size; (void)ws_size;

    char* ws = (char*)d_ws;
    _Float16* hs16   = (_Float16*)(ws);                       //  32 MB: S x 8192 f16
    float*    qkv32  = (float*)  (ws + ((size_t)32  << 20));  //  48 MB: S x 6144 f32
    _Float16* q16    = (_Float16*)(ws + ((size_t)80  << 20)); //  16 MB: 32 x S x 128
    _Float16* k16    = (_Float16*)(ws + ((size_t)96  << 20)); //   4 MB: 8 x S x 128
    _Float16* v16    = (_Float16*)(ws + ((size_t)100 << 20)); //   4 MB
    _Float16* attn16 = (_Float16*)(ws + ((size_t)104 << 20)); //  16 MB: S x 4096

    const int nhs = S_LEN * HID2;
    f32_to_f16_kernel<<<nhs / 1024, 256, 0, stream>>>(hs, hs16, nhs);

    dim3 gq(4096 / 128, S_LEN / 128);
    gemm_f16w32_kernel<<<gq, 256, 0, stream>>>(hs16, Wq, qkv32,        HID2, QKV_COLS);
    dim3 gk(1024 / 128, S_LEN / 128);
    gemm_f16w32_kernel<<<gk, 256, 0, stream>>>(hs16, Wk, qkv32 + 4096, HID2, QKV_COLS);
    gemm_f16w32_kernel<<<gk, 256, 0, stream>>>(hs16, Wv, qkv32 + 5120, HID2, QKV_COLS);

    const int nrope = S_LEN * 40 * 64;
    rope_qk_kernel<<<nrope / 256, 256, 0, stream>>>(qkv32, fr, q16, k16);
    const int nv = S_LEN * 1024;
    conv_v_kernel<<<nv / 1024, 256, 0, stream>>>(qkv32, v16);

    dim3 ga(S_LEN / 64, NH);
    attn_kernel<<<ga, 128, 0, stream>>>(q16, k16, v16, ckp, cvp, attn16);

    dim3 go(4096 / 128, S_LEN / 128);
    gemm_f16w32_kernel<<<go, 256, 0, stream>>>(attn16, Wo, out, 4096, 4096);
}
